// MultiHeadSelfAttention_11501922419147
// MI455X (gfx1250) — matrix operand order repaired and checked by reading, not compiled, not run on hardware
//
#include <hip/hip_runtime.h>

typedef __bf16 bf16_t;
typedef __bf16 v16bf __attribute__((ext_vector_type(16)));
typedef __bf16 v8bf  __attribute__((ext_vector_type(8)));
typedef float  v8f   __attribute__((ext_vector_type(8)));
typedef float  v4f   __attribute__((ext_vector_type(4)));
typedef int    v4i   __attribute__((ext_vector_type(4)));

typedef __attribute__((address_space(1))) v4i* gv4p;   // global v4i*
typedef __attribute__((address_space(3))) v4i* sv4p;   // LDS v4i*

#define D_MODEL 1024
#define SEQ     2048
#define BATCH   2
#define HEADS   16
#define DH      64
#define MTOT    (BATCH * SEQ)   // 4096

#define WMMA_BF16(a, b, c) \
  __builtin_amdgcn_wmma_f32_16x16x32_bf16(false, (a), false, (b), (short)0, (c), false, false)

static __device__ __forceinline__ bf16_t f2bf(float f) {
  unsigned u = __builtin_bit_cast(unsigned, f);
  unsigned r = u + 0x7FFFu + ((u >> 16) & 1u);   // round-to-nearest-even
  unsigned short h = (unsigned short)(r >> 16);
  return __builtin_bit_cast(bf16_t, h);
}

static __device__ __forceinline__ v16bf cat8(v8bf lo, v8bf hi) {
  return __builtin_shufflevector(lo, hi, 0, 1, 2, 3, 4, 5, 6, 7,
                                         8, 9, 10, 11, 12, 13, 14, 15);
}

// ---------------------------------------------------------------------------
// fp32 -> bf16 cast, 4 elements/thread
// ---------------------------------------------------------------------------
__global__ void __launch_bounds__(256)
cvt_f32_bf16(const float* __restrict__ in, bf16_t* __restrict__ out, int n) {
  int i = (blockIdx.x * 256 + threadIdx.x) * 4;
  if (i + 3 < n) {
    v4f v = *(const v4f*)(in + i);
    union { bf16_t h[4]; uint2 u; } o;
    o.h[0] = f2bf(v.x); o.h[1] = f2bf(v.y);
    o.h[2] = f2bf(v.z); o.h[3] = f2bf(v.w);
    *(uint2*)(out + i) = o.u;
  }
}

// ---------------------------------------------------------------------------
// Async copy of a 128x32 bf16 tile (row stride K in global, 32 in LDS).
// 256 threads x 2 chunks x 16B = 8 KB, via GLOBAL_LOAD_ASYNC_TO_LDS_B128
// (no VGPR data path, tracked by ASYNCcnt, overlaps WMMA compute).
// ---------------------------------------------------------------------------
static __device__ __forceinline__ void async_cp_tile(
    const bf16_t* __restrict__ gsrc, bf16_t* lds, int K, int rowBase, int k0) {
  const int t = threadIdx.x;
#pragma unroll
  for (int j = 0; j < 2; ++j) {
    int c   = t + 256 * j;        // 0..511 16B chunks
    int row = c >> 2;
    int ko  = (c & 3) * 8;        // element offset within row
    const bf16_t* g = gsrc + (size_t)(rowBase + row) * K + k0 + ko;
    bf16_t*       l = lds + row * 32 + ko;
    __builtin_amdgcn_global_load_async_to_lds_b128(
        (gv4p)g, (sv4p)l, /*imm offset*/ 0, /*cpol*/ 0);
  }
}

// ---------------------------------------------------------------------------
// One K-step: read A/B fragments from an LDS tile and do 8 WMMAs (32x64/wave).
// A-fragment per-lane K pattern: {half*8 + 0..7, half*8 + 16..23}
// B-fragment per-lane K pattern: {half*8 + 0..7, half*8 + 16..23}
// ---------------------------------------------------------------------------
static __device__ __forceinline__ void gemm_step(
    const bf16_t* sAbuf, const bf16_t* sBbuf, int mloc, int nloc,
    int l15, int half, v8f (&acc)[2][4]) {
  const bf16_t* a0p = sAbuf + (mloc + l15) * 32 + half * 8;
  const bf16_t* a1p = sAbuf + (mloc + 16 + l15) * 32 + half * 8;
  v16bf a0 = cat8(*(const v8bf*)(a0p), *(const v8bf*)(a0p + 16));
  v16bf a1 = cat8(*(const v8bf*)(a1p), *(const v8bf*)(a1p + 16));
#pragma unroll
  for (int j = 0; j < 4; ++j) {
    const bf16_t* bp = sBbuf + (nloc + 16 * j + l15) * 32 + half * 8;
    v16bf b = cat8(*(const v8bf*)(bp), *(const v8bf*)(bp + 16));
    acc[0][j] = WMMA_BF16(a0, b, acc[0][j]);
    acc[1][j] = WMMA_BF16(a1, b, acc[1][j]);
  }
}

// ---------------------------------------------------------------------------
// Wave-level GEMM core with double-buffered async-LDS staging.
// C(32x64 per wave) += A[mBase..,k] * W^T ; W is [N,K] row-major, so
// B[k,n] = W[n,k] (B-fragment lane n reads row n -> contiguous).
// 8 waves/block -> 128x128 block tile, K-step 32. Last step peeled so the
// steady-state loop body is branch-free.
// ---------------------------------------------------------------------------
static __device__ __forceinline__ void gemm_core_lds(
    const bf16_t* __restrict__ A, const bf16_t* __restrict__ Wt,
    int K, int mBase, int nBase, bf16_t* sA, bf16_t* sB, v8f (&acc)[2][4]) {
  const int lane = threadIdx.x & 31;
  const int w    = threadIdx.x >> 5;
  const int l15  = lane & 15, half = lane >> 4;
  const int mloc = (w & 3) * 32;
  const int nloc = (w >> 2) * 64;

  async_cp_tile(A,  sA, K, mBase, 0);
  async_cp_tile(Wt, sB, K, nBase, 0);

  const int nk = K / 32;
  for (int i = 0; i < nk - 1; ++i) {
    // own async loads for tile i complete; barrier also guarantees everyone
    // is done READING the buffer we are about to overwrite (used in iter i-1)
    asm volatile("s_wait_asynccnt 0x0" ::: "memory");
    __syncthreads();
    const int cur = (i & 1) * (128 * 32);
    const int nxt = cur ^ (128 * 32);
    async_cp_tile(A,  sA + nxt, K, mBase, (i + 1) * 32);
    async_cp_tile(Wt, sB + nxt, K, nBase, (i + 1) * 32);
    gemm_step(sA + cur, sB + cur, mloc, nloc, l15, half, acc);
  }
  asm volatile("s_wait_asynccnt 0x0" ::: "memory");
  __syncthreads();
  const int last = ((nk - 1) & 1) * (128 * 32);
  gemm_step(sA + last, sB + last, mloc, nloc, l15, half, acc);
}

// ---------------------------------------------------------------------------
// QKV projection: y = x @ W.T ; blockIdx.z selects Q/K/V.
// Q,K stored [B,H,S,Dh]; V stored transposed [B,H,Dh,S]. Q pre-scaled 1/8.
// ---------------------------------------------------------------------------
__global__ void __launch_bounds__(256)
qkv_gemm(const bf16_t* __restrict__ xh,
         const bf16_t* __restrict__ Wq, const bf16_t* __restrict__ Wk,
         const bf16_t* __restrict__ Wv,
         bf16_t* __restrict__ Qb, bf16_t* __restrict__ Kb,
         bf16_t* __restrict__ Vtb) {
  __shared__ bf16_t sA[2 * 128 * 32];   // 16 KB
  __shared__ bf16_t sB[2 * 128 * 32];   // 16 KB

  const int mBase = blockIdx.y * 128;
  const int nBase = blockIdx.x * 128;
  const int mode  = blockIdx.z;
  const bf16_t* Wt = (mode == 0) ? Wq : (mode == 1) ? Wk : Wv;
  bf16_t* dst = (mode == 0) ? Qb : (mode == 1) ? Kb : Vtb;
  const float oscale = (mode == 0) ? 0.125f : 1.0f;   // 1/sqrt(Dh) folded into Q

  v8f acc[2][4] = {};
  gemm_core_lds(xh, Wt, D_MODEL, mBase, nBase, sA, sB, acc);

  const int w = threadIdx.x >> 5;
  const int m0 = mBase + (w & 3) * 32;
  const int n0 = nBase + (w >> 2) * 64;
  const int lane = threadIdx.x & 31;
  const int l15 = lane & 15, half = lane >> 4;
#pragma unroll
  for (int i = 0; i < 2; ++i)
#pragma unroll
    for (int j = 0; j < 4; ++j)
#pragma unroll
      for (int r = 0; r < 8; ++r) {
        int mg = m0 + 16 * i + 8 * half + r;     // row of x = b*S + s
        int ng = n0 + 16 * j + l15;              // column = h*Dh + dh
        int b = mg >> 11, s = mg & 2047;
        int h = ng >> 6,  dh = ng & 63;
        bf16_t hv = f2bf(acc[i][j][r] * oscale);
        size_t addr = (mode == 2)
            ? (((size_t)(b * HEADS + h) * DH + dh) * SEQ + s)     // V transposed
            : (((size_t)(b * HEADS + h) * SEQ + s) * DH + dh);    // Q,K
        dst[addr] = hv;
      }
}

// ---------------------------------------------------------------------------
// Flash attention with causal mask. Block = (b,h, 128 q rows); wave = 16 rows.
// Streams 32-key blocks: QK^T (4 wmma) -> online softmax -> P via LDS -> PV (4 wmma)
// ---------------------------------------------------------------------------
__global__ void __launch_bounds__(256)
flash_attn(const bf16_t* __restrict__ Qb, const bf16_t* __restrict__ Kb,
           const bf16_t* __restrict__ Vtb, bf16_t* __restrict__ Ob) {
  __shared__ bf16_t pbuf[8][16 * 32];   // per-wave 16x32 P tile (8 KB)

  const int w    = threadIdx.x >> 5;
  const int lane = threadIdx.x & 31;
  const int l15  = lane & 15, half = lane >> 4;
  const int bh   = blockIdx.x;                 // b*H + h
  const int qBase = blockIdx.y * 128 + w * 16;

  const bf16_t* Qp = Qb  + (size_t)bh * SEQ * DH;
  const bf16_t* Kp = Kb  + (size_t)bh * SEQ * DH;
  const bf16_t* Vp = Vtb + (size_t)bh * DH * SEQ;

  // Q A-fragments for d = 0..31 and 32..63 (held in registers for whole loop)
  v16bf aq[2];
  {
    const bf16_t* qr = Qp + (size_t)(qBase + l15) * DH + half * 8;
#pragma unroll
    for (int t = 0; t < 2; ++t)
      aq[t] = cat8(*(const v8bf*)(qr + 32 * t), *(const v8bf*)(qr + 32 * t + 16));
  }

  float mrow[8], lrow[8];
  v8f oacc[4] = {};
#pragma unroll
  for (int r = 0; r < 8; ++r) { mrow[r] = -1e30f; lrow[r] = 0.0f; }

  const int kEnd = qBase + 16;                 // causal: keys <= last q row
  for (int kb = 0; kb < kEnd; kb += 32) {
    // ---- scores: two 16x16 C tiles over 32 keys, reduce over d (2 wmma each)
    v8f s[2] = {};
#pragma unroll
    for (int j = 0; j < 2; ++j) {
      const bf16_t* kr = Kp + (size_t)(kb + 16 * j + l15) * DH + half * 8;
#pragma unroll
      for (int t = 0; t < 2; ++t) {
        v16bf bk = cat8(*(const v8bf*)(kr + 32 * t), *(const v8bf*)(kr + 32 * t + 16));
        s[j] = WMMA_BF16(aq[t], bk, s[j]);
      }
    }

    // ---- causal mask + online softmax (rows live across 16-lane halves)
    float scl[8];
#pragma unroll
    for (int r = 0; r < 8; ++r) {
      const int q  = qBase + 8 * half + r;
      const int k0 = kb + l15, k1 = kb + 16 + l15;
      float s0 = (k0 <= q) ? s[0][r] : -1e30f;
      float s1 = (k1 <= q) ? s[1][r] : -1e30f;
      float mx = fmaxf(s0, s1);
      mx = fmaxf(mx, __shfl_xor(mx, 1));
      mx = fmaxf(mx, __shfl_xor(mx, 2));
      mx = fmaxf(mx, __shfl_xor(mx, 4));
      mx = fmaxf(mx, __shfl_xor(mx, 8));
      float mnew = fmaxf(mrow[r], mx);
      float p0 = __expf(s0 - mnew);
      float p1 = __expf(s1 - mnew);
      float rs = p0 + p1;
      rs += __shfl_xor(rs, 1);
      rs += __shfl_xor(rs, 2);
      rs += __shfl_xor(rs, 4);
      rs += __shfl_xor(rs, 8);
      float sc = __expf(mrow[r] - mnew);
      lrow[r] = lrow[r] * sc + rs;
      mrow[r] = mnew;
      scl[r]  = sc;
      // stash P (C layout) into this wave's LDS tile
      pbuf[w][(8 * half + r) * 32 + l15]      = f2bf(p0);
      pbuf[w][(8 * half + r) * 32 + 16 + l15] = f2bf(p1);
    }
#pragma unroll
    for (int f = 0; f < 4; ++f)
#pragma unroll
      for (int r = 0; r < 8; ++r) oacc[f][r] *= scl[r];

    // DS ops from one wave are in-order; wait for stores, then re-read in A layout
    asm volatile("s_wait_dscnt 0x0" ::: "memory");
    v16bf pa;
    {
      const bf16_t* pr = &pbuf[w][l15 * 32 + half * 8];
      pa = cat8(*(const v8bf*)(pr), *(const v8bf*)(pr + 16));
    }

    // ---- O += P @ V  (V transposed: row dv contiguous over keys)
#pragma unroll
    for (int f = 0; f < 4; ++f) {
      const bf16_t* vr = Vp + (size_t)(16 * f + l15) * SEQ + kb + half * 8;
      v16bf bv = cat8(*(const v8bf*)(vr), *(const v8bf*)(vr + 16));
      oacc[f] = WMMA_BF16(pa, bv, oacc[f]);
    }
  }

  // ---- epilogue: normalize and store to [B,S,D]
  const int b = bh >> 4, h = bh & 15;
#pragma unroll
  for (int f = 0; f < 4; ++f)
#pragma unroll
    for (int r = 0; r < 8; ++r) {
      int s_idx = qBase + 8 * half + r;
      int dv = 16 * f + l15;
      Ob[(size_t)(b * SEQ + s_idx) * D_MODEL + h * DH + dv] =
          f2bf(oacc[f][r] / lrow[r]);
    }
}

// ---------------------------------------------------------------------------
// Output projection: out = O @ Wo.T  (fp32 result)
// ---------------------------------------------------------------------------
__global__ void __launch_bounds__(256)
proj_gemm(const bf16_t* __restrict__ Ob, const bf16_t* __restrict__ Woh,
          float* __restrict__ out) {
  __shared__ bf16_t sA[2 * 128 * 32];   // 16 KB
  __shared__ bf16_t sB[2 * 128 * 32];   // 16 KB

  const int mBase = blockIdx.y * 128;
  const int nBase = blockIdx.x * 128;

  v8f acc[2][4] = {};
  gemm_core_lds(Ob, Woh, D_MODEL, mBase, nBase, sA, sB, acc);

  const int w = threadIdx.x >> 5;
  const int m0 = mBase + (w & 3) * 32;
  const int n0 = nBase + (w >> 2) * 64;
  const int lane = threadIdx.x & 31;
  const int l15 = lane & 15, half = lane >> 4;
#pragma unroll
  for (int i = 0; i < 2; ++i)
#pragma unroll
    for (int j = 0; j < 4; ++j)
#pragma unroll
      for (int r = 0; r < 8; ++r) {
        int mg = m0 + 16 * i + 8 * half + r;
        int ng = n0 + 16 * j + l15;
        out[(size_t)mg * D_MODEL + ng] = acc[i][j][r];
      }
}

// ---------------------------------------------------------------------------
extern "C" void kernel_launch(void* const* d_in, const int* in_sizes, int n_in,
                              void* d_out, int out_size, void* d_ws, size_t ws_size,
                              hipStream_t stream) {
  (void)in_sizes; (void)n_in; (void)out_size; (void)ws_size;
  const float* x  = (const float*)d_in[0];
  const float* Wq = (const float*)d_in[1];
  const float* Wk = (const float*)d_in[2];
  const float* Wv = (const float*)d_in[3];
  const float* Wo = (const float*)d_in[4];

  char* ws = (char*)d_ws;
  const size_t MB = 1u << 20;
  bf16_t* xh  = (bf16_t*)(ws + 0 * MB);    // 8 MB  [4096,1024]
  bf16_t* Wqh = (bf16_t*)(ws + 8 * MB);    // 2 MB
  bf16_t* Wkh = (bf16_t*)(ws + 10 * MB);
  bf16_t* Wvh = (bf16_t*)(ws + 12 * MB);
  bf16_t* Woh = (bf16_t*)(ws + 14 * MB);
  bf16_t* Qb  = (bf16_t*)(ws + 16 * MB);   // 8 MB  [B,H,S,Dh] (pre-scaled)
  bf16_t* Kb  = (bf16_t*)(ws + 24 * MB);   // 8 MB  [B,H,S,Dh]
  bf16_t* Vtb = (bf16_t*)(ws + 32 * MB);   // 8 MB  [B,H,Dh,S] (transposed)
  bf16_t* Ob  = (bf16_t*)(ws + 40 * MB);   // 8 MB  [B,S,D]

  const int nX = MTOT * D_MODEL;           // 4,194,304
  const int nW = D_MODEL * D_MODEL;        // 1,048,576
  cvt_f32_bf16<<<nX / 1024, 256, 0, stream>>>(x,  xh,  nX);
  cvt_f32_bf16<<<nW / 1024, 256, 0, stream>>>(Wq, Wqh, nW);
  cvt_f32_bf16<<<nW / 1024, 256, 0, stream>>>(Wk, Wkh, nW);
  cvt_f32_bf16<<<nW / 1024, 256, 0, stream>>>(Wv, Wvh, nW);
  cvt_f32_bf16<<<nW / 1024, 256, 0, stream>>>(Wo, Woh, nW);

  dim3 gq(D_MODEL / 128, MTOT / 128, 3);   // 8 x 32 x 3
  qkv_gemm<<<gq, 256, 0, stream>>>(xh, Wqh, Wkh, Wvh, Qb, Kb, Vtb);

  dim3 ga(BATCH * HEADS, SEQ / 128, 1);    // 32 x 16
  flash_attn<<<ga, 256, 0, stream>>>(Qb, Kb, Vtb, Ob);

  dim3 gp(D_MODEL / 128, MTOT / 128, 1);   // 8 x 32
  proj_gemm<<<gp, 256, 0, stream>>>(Ob, Woh, (float*)d_out);
}
